// Attention_18794776887711
// MI455X (gfx1250) — matrix operand order repaired and checked by reading, not compiled, not run on hardware
//
#include <hip/hip_runtime.h>
#include <hip/hip_bf16.h>
#include <math.h>

// Problem constants (from reference): B=8, TQ=TK=256, DQ=DK=DV=512, H=256
#define BQ   8
#define TT   256
#define DD   512
#define HH   256

typedef __attribute__((ext_vector_type(16))) __bf16 v16bf;
typedef __attribute__((ext_vector_type(8)))  float  v8f;

#if __has_builtin(__builtin_amdgcn_tanhf)
#define DEV_TANH(x) __builtin_amdgcn_tanhf(x)
#else
#define DEV_TANH(x) tanhf(x)
#endif

// ---------------------------------------------------------------------------
// Kernel 1: projections  qp = Q@Wq, kp = K@Wk   (each a 2048x512 @ 512x256 GEMM)
// One wave per 16x16 output tile, K swept 32 at a time with bf16 WMMA.
// 4096 tiles total (2048 for Q, 2048 for K); 8 waves per 256-thread block.
// ---------------------------------------------------------------------------
__global__ __launch_bounds__(256) void proj_wmma_kernel(
    const float* __restrict__ queries, const float* __restrict__ keys,
    const float* __restrict__ Wq, const float* __restrict__ Wk,
    float* __restrict__ qp, float* __restrict__ kp)
{
    const int lane = threadIdx.x & 31;
    const int wave = threadIdx.x >> 5;
    int g = blockIdx.x * 8 + wave;                 // 0..4095 tile id

    const float* X; const float* W; float* P;
    if (g < 2048) { X = queries; W = Wq; P = qp; }
    else          { g -= 2048; X = keys; W = Wk; P = kp; }

    const int mtile = g >> 4;                      // 0..127 (rows of flattened B*T)
    const int ntile = g & 15;                      // 0..15  (H/16)
    const int mrow  = lane & 15;
    const int hiA   = (lane >> 4) * 16;            // K sub-offset for A (contiguous k = 16h + s order)
    const int hiB   = (lane >> 4) * 16;            // K sub-offset for B (contiguous order, same as A)
    const int ncol  = ntile * 16 + mrow;           // B/N column owned by this lane

    const float* Xrow = X + (size_t)(mtile * 16 + mrow) * DD;

    v8f acc = {};
    for (int kk = 0; kk < DD; kk += 32) {
        v16bf a, bm;
        // A (16x32 bf16): lane<16 -> K kk..kk+15; lane>=16 -> +16
        #pragma unroll
        for (int i = 0; i < 8; ++i) {
            a[i]     = (__bf16)Xrow[kk + hiA + i];
            a[i + 8] = (__bf16)Xrow[kk + 8 + hiA + i];
        }
        // B (32x16 bf16): lane holds one column, 16 consecutive K values (+16 for hi half)
        #pragma unroll
        for (int i = 0; i < 16; ++i) {
            bm[i] = (__bf16)W[(size_t)(kk + hiB + i) * HH + ncol];
        }
        acc = __builtin_amdgcn_wmma_f32_16x16x32_bf16(
            false, a, false, bm, (short)0, acc, false, false);
    }

    // C layout: VGPR i -> row (i + 8*(lane>=16)), col = ncol
    const int mofs = (lane >> 4) * 8;
    #pragma unroll
    for (int i = 0; i < 8; ++i) {
        P[(size_t)(mtile * 16 + mofs + i) * HH + ncol] = acc[i];
    }
}

// ---------------------------------------------------------------------------
// Kernel 2: fused additive-attention scores + softmax + attn@V
// One 256-thread workgroup per (batch b, 16-query tile).
// Each thread owns one k column: scores kept in 16 registers, softmax via LDS
// reductions, unnormalized exp matrix in LDS feeds bf16 WMMA for attn@V, and
// 1/rowsum is folded into the output store.
// ---------------------------------------------------------------------------
__global__ __launch_bounds__(256) void fused_attn_kernel(
    const float* __restrict__ qp, const float* __restrict__ kp,
    const float* __restrict__ values, const float* __restrict__ wv,
    float* __restrict__ out)
{
    __shared__ __align__(16) float lds_q[16 * HH];    // 16 KB: q-projection tile
    __shared__ __align__(16) float lds_e[16 * HH];    // 16 KB: scores -> exp(scores-max)
    __shared__ __align__(16) float lds_wv[HH];        // 1 KB
    __shared__ float lds_red[16 * 16];
    __shared__ float lds_rmax[16];
    __shared__ float lds_rinv[16];

    const int tid = threadIdx.x;
    const int b   = blockIdx.x >> 4;                  // 8 batches * 16 q-tiles = 128 blocks
    const int qt  = blockIdx.x & 15;
    const int q0  = qt * 16;
    const size_t qbase = ((size_t)b * TT + q0) * HH;  // 4096 contiguous floats

    for (int i = tid; i < 16 * HH; i += 256) lds_q[i] = qp[qbase + i];
    lds_wv[tid] = wv[tid];
    __syncthreads();

    // ---- score loop: thread owns k = tid; acc[q] = sum_h wv[h]*tanh(qp+kp) ----
    const int k = tid;
    const float* kprow = kp + ((size_t)b * TT + k) * HH;
    float acc[16];
    #pragma unroll
    for (int q = 0; q < 16; ++q) acc[q] = 0.0f;

    for (int h = 0; h < HH; h += 4) {
        const float4 kc = *(const float4*)(kprow + h);
        const float4 w4 = *(const float4*)(lds_wv + h);
        #pragma unroll
        for (int q = 0; q < 16; ++q) {
            const float4 qv = *(const float4*)(lds_q + q * HH + h);  // LDS broadcast
            acc[q] += w4.x * DEV_TANH(qv.x + kc.x);
            acc[q] += w4.y * DEV_TANH(qv.y + kc.y);
            acc[q] += w4.z * DEV_TANH(qv.z + kc.z);
            acc[q] += w4.w * DEV_TANH(qv.w + kc.w);
        }
    }

    // ---- softmax over k (row-wise) ----
    #pragma unroll
    for (int q = 0; q < 16; ++q) lds_e[q * HH + k] = acc[q];
    __syncthreads();
    {   // partial max: 16 threads per row
        const int row = tid >> 4, j = tid & 15;
        float m = -__builtin_inff();
        #pragma unroll 4
        for (int i = 0; i < 16; ++i) m = fmaxf(m, lds_e[row * HH + j * 16 + i]);
        lds_red[row * 16 + j] = m;
    }
    __syncthreads();
    if (tid < 16) {
        float m = -__builtin_inff();
        #pragma unroll 4
        for (int i = 0; i < 16; ++i) m = fmaxf(m, lds_red[tid * 16 + i]);
        lds_rmax[tid] = m;
    }
    __syncthreads();
    #pragma unroll
    for (int q = 0; q < 16; ++q) {
        acc[q] = __expf(acc[q] - lds_rmax[q]);
        lds_e[q * HH + k] = acc[q];                   // unnormalized attn weights
    }
    __syncthreads();
    {   // partial sum
        const int row = tid >> 4, j = tid & 15;
        float s = 0.0f;
        #pragma unroll 4
        for (int i = 0; i < 16; ++i) s += lds_e[row * HH + j * 16 + i];
        lds_red[row * 16 + j] = s;
    }
    __syncthreads();
    if (tid < 16) {
        float s = 0.0f;
        #pragma unroll 4
        for (int i = 0; i < 16; ++i) s += lds_red[tid * 16 + i];
        lds_rinv[tid] = 1.0f / s;
    }
    __syncthreads();

    // ---- attn@V: out[16,512] = e[16,256] @ V[b][256,512], bf16 WMMA, scale by rinv ----
    const int lane = tid & 31;
    const int wave = tid >> 5;
    const int mrow = lane & 15;
    const int hiA  = (lane >> 4) * 16;
    const int hiB  = (lane >> 4) * 16;
    const int mofs = (lane >> 4) * 8;
    const float* Vb = values + (size_t)b * TT * DD;

    for (int nt = wave; nt < DD / 16; nt += 8) {      // 32 n-tiles, 8 waves
        const int ncol = nt * 16 + mrow;
        v8f c = {};
        for (int kk = 0; kk < TT; kk += 32) {
            v16bf a, bm;
            #pragma unroll
            for (int i = 0; i < 8; ++i) {
                a[i]     = (__bf16)lds_e[mrow * HH + kk + hiA + i];
                a[i + 8] = (__bf16)lds_e[mrow * HH + kk + 8 + hiA + i];
            }
            #pragma unroll
            for (int i = 0; i < 16; ++i) {
                bm[i] = (__bf16)Vb[(size_t)(kk + hiB + i) * DD + ncol];
            }
            c = __builtin_amdgcn_wmma_f32_16x16x32_bf16(
                false, a, false, bm, (short)0, c, false, false);
        }
        #pragma unroll
        for (int i = 0; i < 8; ++i) {
            const int m = mofs + i;
            out[((size_t)b * TT + q0 + m) * DD + ncol] = c[i] * lds_rinv[m];
        }
    }
}

// ---------------------------------------------------------------------------
extern "C" void kernel_launch(void* const* d_in, const int* in_sizes, int n_in,
                              void* d_out, int out_size, void* d_ws, size_t ws_size,
                              hipStream_t stream) {
    (void)in_sizes; (void)n_in; (void)out_size; (void)ws_size;
    const float* queries = (const float*)d_in[0];   // (8,256,512)
    const float* keys    = (const float*)d_in[1];   // (8,256,512)
    const float* values  = (const float*)d_in[2];   // (8,256,512)
    const float* Wq      = (const float*)d_in[3];   // (512,256)
    const float* Wk      = (const float*)d_in[4];   // (512,256)
    const float* wv      = (const float*)d_in[5];   // (256,)

    float* qp = (float*)d_ws;                       // (2048,256) = 2 MB
    float* kp = qp + (size_t)BQ * TT * HH;          // (2048,256) = 2 MB

    proj_wmma_kernel<<<512, 256, 0, stream>>>(queries, keys, Wq, Wk, qp, kp);
    fused_attn_kernel<<<BQ * (TT / 16), 256, 0, stream>>>(qp, kp, values, wv,
                                                          (float*)d_out);
}
